// Decoder_65377992180424
// MI455X (gfx1250) — matrix operand order repaired and checked by reading, not compiled, not run on hardware
//
#include <hip/hip_runtime.h>
#include <hip/hip_bf16.h>
#include <cstdint>

// ---------------------------------------------------------------------------
// Decoder forward (node + edge transformer) for MI455X / gfx1250, wave32.
// GEMMs (K=256, N in {256,512,768}) use v_wmma_f32_16x16x32_f16 with:
//   - A tiles DMA'd global->LDS via global_load_async_to_lds_b128 (ASYNCcnt)
//   - weights pre-converted to f16 once per launch (direct v16h B fragments)
//   - each wave computing two 16x16 N-tiles off one shared A fragment.
// Attention masks are exploited structurally (causal / group-diagonal /
// 1-key / 2-key), numerically identical to the -1e9 additive masks.
// Workspace: ~83 MB f32 scratch from d_ws.
// ---------------------------------------------------------------------------

typedef __attribute__((ext_vector_type(16))) _Float16 v16h;
typedef __attribute__((ext_vector_type(8)))  float    v8f;

static constexpr int S   = 256;
static constexpr int Bz  = 4;
static constexpr int H   = 256;
static constexpr int LE  = 2994;      // number of edge queries
static constexpr int LEB = LE * Bz;   // 11976 edge rows
static constexpr float RSQ_DH = 0.08838834764831845f;  // 1/sqrt(128)

// ---------------------------------------------------------------------------
// f32 -> f16 weight conversion (once per launch; weights are reused ~750x).
// ---------------------------------------------------------------------------
__global__ void k_cvt(const float* __restrict__ s, _Float16* __restrict__ d, int n) {
  int i = blockIdx.x * 256 + threadIdx.x;
  if (i < n) d[i] = (_Float16)s[i];
}

// ---------------------------------------------------------------------------
// WMMA GEMM:  C[M,N] = act(A[M,256] @ W[N,256]^T + bias[N]),  W in f16.
// grid = (N/256, ceil(M/16)); block = 256 threads = 8 waves.
// Wave w owns the two 16x16 tiles at cols blockIdx.x*256 + w*32 (+16).
// A tile is staged to LDS by async DMA (GLOBAL_LOAD_ASYNC_TO_LDS_B128).
// ---------------------------------------------------------------------------
__global__ __launch_bounds__(256) void k_gemm(const float* __restrict__ A,
                                              const _Float16* __restrict__ W,
                                              const float* __restrict__ bias,
                                              float* __restrict__ C,
                                              int M, int N, int act) {
  const int K = 256;
  __shared__ float As[16][264];            // 264 stride dodges bank conflicts
  const int tid  = threadIdx.x;
  const int lane = tid & 31;
  const int wave = tid >> 5;
  const int mbase = blockIdx.y * 16;
  const int nbase = blockIdx.x * 256 + wave * 32;

  // ---- Async DMA the 16x256 f32 A tile into LDS (GV mode, b128/lane). ----
  // Rows >= M are simply not fetched: they only feed C rows that are never
  // stored (WMMA output row m depends only on A row m).
  {
    unsigned lbase = (unsigned)(uintptr_t)&As[0][0];
#pragma unroll
    for (int it = 0; it < 4; ++it) {
      int idx = tid + it * 256;
      int r = idx >> 6;                  // row within tile
      int cf = (idx & 63) * 4;           // f32 column (16B chunks)
      if (mbase + r < M) {
        unsigned laddr = lbase + (unsigned)(r * 264 + cf) * 4u;
        unsigned long long ga =
            (unsigned long long)(uintptr_t)(A + (size_t)(mbase + r) * K + cf);
        asm volatile("global_load_async_to_lds_b128 %0, %1, off"
                     :: "v"(laddr), "v"(ga) : "memory");
      }
    }
    asm volatile("s_wait_asynccnt 0x0" ::: "memory");
  }
  __syncthreads();

  // Fragment geometry (ISA 7.12.2, 16-bit A 16x32 / B 32x16, wave32).
  const int mlane = lane & 15;
  const int ahalf = (lane >> 4) * 8;     // A: lanes<16 -> K 0..7/16..23, else 8..15/24..31
  const int n0 = nbase + mlane;          // B/C/D column = lane%16
  const int n1 = n0 + 16;
  const int bko = (lane >> 4) * 8;       // B: lanes<16 -> K 0..7/16..23, else 8..15/24..31
  const _Float16* w0 = W + (size_t)n0 * K;
  const _Float16* w1 = W + (size_t)n1 * K;

  v8f acc0 = {0.f, 0.f, 0.f, 0.f, 0.f, 0.f, 0.f, 0.f};
  v8f acc1 = {0.f, 0.f, 0.f, 0.f, 0.f, 0.f, 0.f, 0.f};
#pragma unroll 2
  for (int kb = 0; kb < K; kb += 32) {
    v16h af;
#pragma unroll
    for (int v = 0; v < 4; ++v) {
      int k1 = kb + ahalf + 2 * v;       // VGPR v   : K pair
      int k2 = k1 + 16;                  // VGPR v+4 : K pair
      af[2 * v]     = (_Float16)As[mlane][k1];
      af[2 * v + 1] = (_Float16)As[mlane][k1 + 1];
      af[8 + 2 * v]     = (_Float16)As[mlane][k2];
      af[8 + 2 * v + 1] = (_Float16)As[mlane][k2 + 1];
    }
    if (kb + 32 < K) __builtin_prefetch(w0 + kb + 32, 0, 0);
    typedef __attribute__((ext_vector_type(8))) _Float16 v8h_;
    v8h_ b0l = *(const v8h_*)(w0 + kb + bko), b0h = *(const v8h_*)(w0 + kb + 16 + bko);
    v8h_ b1l = *(const v8h_*)(w1 + kb + bko), b1h = *(const v8h_*)(w1 + kb + 16 + bko);
    v16h bf0 = __builtin_shufflevector(b0l, b0h, 0,1,2,3,4,5,6,7,8,9,10,11,12,13,14,15);
    v16h bf1 = __builtin_shufflevector(b1l, b1h, 0,1,2,3,4,5,6,7,8,9,10,11,12,13,14,15);
    acc0 = __builtin_amdgcn_wmma_f32_16x16x32_f16(false, af, false, bf0,
                                                  (short)0, acc0, false, false);
    acc1 = __builtin_amdgcn_wmma_f32_16x16x32_f16(false, af, false, bf1,
                                                  (short)0, acc1, false, false);
  }

  // C/D layout: lanes<16 -> rows mbase+0..7, lanes>=16 -> rows mbase+8..15.
  const float bv0 = bias ? bias[n0] : 0.f;
  const float bv1 = bias ? bias[n1] : 0.f;
  const int mrow0 = mbase + ((lane >> 4) * 8);
#pragma unroll
  for (int r = 0; r < 8; ++r) {
    int m = mrow0 + r;
    if (m < M) {
      float v0 = acc0[r] + bv0;
      float v1 = acc1[r] + bv1;
      if (act) { v0 = fmaxf(v0, 0.f); v1 = fmaxf(v1, 0.f); }
      C[(size_t)m * N + n0] = v0;
      C[(size_t)m * N + n1] = v1;
    }
  }
}

static inline void gemm(hipStream_t st, const float* A, const _Float16* W,
                        const float* b, float* C, int M, int N, int act) {
  dim3 g(N / 256, (M + 15) / 16);
  k_gemm<<<g, 256, 0, st>>>(A, W, b, C, M, N, act);
}

// ---------------------------------------------------------------------------
// Input projection: x[1024,256] = atom_x[1024,54] @ Wp^T + bp   (tiny K)
// ---------------------------------------------------------------------------
__global__ void k_proj(const float* __restrict__ ax, const float* __restrict__ Wp,
                       const float* __restrict__ bp, float* __restrict__ x) {
  int row = blockIdx.x, t = threadIdx.x;
  __shared__ float sa[54];
  if (t < 54) sa[t] = ax[row * 54 + t];
  __syncthreads();
  const float* wr = Wp + t * 54;
  float a = bp[t];
  for (int j = 0; j < 54; ++j) a += sa[j] * wr[j];
  x[row * 256 + t] = a;
}

// BatchNorm training stats over the 1024 rows, per column.
__global__ void k_bn_stats(const float* __restrict__ x, float* __restrict__ st) {
  int c = blockIdx.x, t = threadIdx.x;
  __shared__ float s1[256], s2[256];
  float a = 0.f, b = 0.f;
  for (int r = t; r < 1024; r += 256) { float v = x[r * 256 + c]; a += v; b += v * v; }
  s1[t] = a; s2[t] = b; __syncthreads();
  for (int off = 128; off; off >>= 1) {
    if (t < off) { s1[t] += s1[t + off]; s2[t] += s2[t + off]; }
    __syncthreads();
  }
  if (t == 0) { float m = s1[0] / 1024.f; st[c] = m; st[256 + c] = s2[0] / 1024.f - m * m; }
}

// BN normalize + SELU + sinusoidal positional encoding -> tgt.
__global__ void k_bn_selu_pe(const float* __restrict__ x, const float* __restrict__ st,
                             const float* __restrict__ g, const float* __restrict__ bb,
                             const int* __restrict__ atom_i, float* __restrict__ tgt) {
  int row = blockIdx.x, d = threadIdx.x;
  float m = st[d], v = st[256 + d];
  float xn = (x[row * 256 + d] - m) * rsqrtf(v + 1e-5f) * g[d] + bb[d];
  const float scale = 1.0507009873554805f, alpha = 1.6732632423543772f;
  float s = xn > 0.f ? scale * xn : scale * alpha * (__expf(xn) - 1.f);
  float pos = (float)atom_i[row];
  float f = __powf(10000.f, -(float)d * (1.f / 256.f));
  float pe = (d & 1) ? __cosf(pos * f) : __sinf(pos * f);
  tgt[row * 256 + d] = pe + s;
}

// ---------------------------------------------------------------------------
// out = LN(base + add) * g + b  per row of 256.  add==nullptr -> plain LN.
// bcast!=0 -> add is indexed by batch (row % 4)   (node cross-attn path).
// ---------------------------------------------------------------------------
__global__ void k_res_ln(const float* __restrict__ base, const float* __restrict__ add,
                         int bcast, const float* __restrict__ g,
                         const float* __restrict__ bt, float* __restrict__ out) {
  int row = blockIdx.x, t = threadIdx.x;
  __shared__ float red[256];
  float v = base[(size_t)row * 256 + t];
  if (add) v += bcast ? add[(row & 3) * 256 + t] : add[(size_t)row * 256 + t];
  red[t] = v; __syncthreads();
  for (int off = 128; off; off >>= 1) { if (t < off) red[t] += red[t + off]; __syncthreads(); }
  float mean = red[0] * (1.f / 256.f);
  __syncthreads();
  float dv = v - mean;
  red[t] = dv * dv; __syncthreads();
  for (int off = 128; off; off >>= 1) { if (t < off) red[t] += red[t + off]; __syncthreads(); }
  float var = red[0] * (1.f / 256.f);
  out[(size_t)row * 256 + t] = dv * rsqrtf(var + 1e-5f) * g[t] + bt[t];
}

// ---------------------------------------------------------------------------
// Node self-attention, causal, S=256, DH=128, 2 heads.
// grid = (S, B*2), block = 256.  qkv rows are s*B+b, cols [q|k|v] of 256 each.
// ---------------------------------------------------------------------------
__global__ void k_node_self(const float* __restrict__ qkv, float* __restrict__ out) {
  int q = blockIdx.x, bh = blockIdx.y;
  int b = bh >> 1, h = bh & 1;
  int t = threadIdx.x;
  __shared__ float sq[128], sc[256], red[256];
  int qrow = q * 4 + b;
  if (t < 128) sq[t] = qkv[(size_t)qrow * 768 + h * 128 + t];
  __syncthreads();
  float s = -1e30f;
  if (t <= q) {
    const float* kp = qkv + (size_t)(t * 4 + b) * 768 + 256 + h * 128;
    float a = 0.f;
    for (int d = 0; d < 128; ++d) a += sq[d] * kp[d];
    s = a * RSQ_DH;
  }
  red[t] = s; __syncthreads();
  for (int off = 128; off; off >>= 1) { if (t < off) red[t] = fmaxf(red[t], red[t + off]); __syncthreads(); }
  float mx = red[0]; __syncthreads();
  float p = (t <= q) ? __expf(s - mx) : 0.f;
  sc[t] = p; red[t] = p; __syncthreads();
  for (int off = 128; off; off >>= 1) { if (t < off) red[t] += red[t + off]; __syncthreads(); }
  float inv = 1.f / red[0];
  __syncthreads();
  if (t < 128) {
    float o = 0.f;
    for (int k = 0; k <= q; ++k)
      o += sc[k] * qkv[(size_t)(k * 4 + b) * 768 + 512 + h * 128 + t];
    out[(size_t)qrow * 256 + h * 128 + t] = o * inv;
  }
}

// Map flat edge index e -> (group, source node index).
__device__ __forceinline__ void edge_decode(int e, int& grp, int& nidx) {
  if (e < 78) {                       // groups 1..12, sizes 1..12
    int g = 1;
    while ((g * (g + 1)) / 2 <= e) ++g;
    grp = g; nidx = e - (g * (g - 1)) / 2;
  } else {                            // groups 13..255, size 12
    int q = (e - 78) / 12, r = (e - 78) % 12;
    grp = 13 + q; nidx = grp - 12 + r;
  }
}

// ---------------------------------------------------------------------------
// Edge self-attention: groups of size min(12,g) attend among themselves.
// grid = (255, B), block = 128; heads looped inside.
// ---------------------------------------------------------------------------
__global__ void k_edge_self(const float* __restrict__ qkv, float* __restrict__ out) {
  int g = blockIdx.x + 1, b = blockIdx.y;
  int sz = g < 12 ? g : 12;
  int estart = (g <= 12) ? (g * (g - 1)) / 2 : 78 + (g - 13) * 12;
  int t = threadIdx.x;
  __shared__ float sq[12][128], sk[12][128], sv[12][128], ss[12][12];
  for (int h = 0; h < 2; ++h) {
    for (int idx = t; idx < sz * 128; idx += 128) {
      int j = idx >> 7, d = idx & 127;
      size_t row = (size_t)((estart + j) * 4 + b) * 768;
      sq[j][d] = qkv[row + h * 128 + d];
      sk[j][d] = qkv[row + 256 + h * 128 + d];
      sv[j][d] = qkv[row + 512 + h * 128 + d];
    }
    __syncthreads();
    for (int p = t; p < sz * sz; p += 128) {
      int j = p / sz, l = p % sz;
      float a = 0.f;
      for (int d = 0; d < 128; ++d) a += sq[j][d] * sk[l][d];
      ss[j][l] = a * RSQ_DH;
    }
    __syncthreads();
    if (t < sz) {
      float mx = -1e30f;
      for (int l = 0; l < sz; ++l) mx = fmaxf(mx, ss[t][l]);
      float sm = 0.f;
      for (int l = 0; l < sz; ++l) { float e = __expf(ss[t][l] - mx); ss[t][l] = e; sm += e; }
      float inv = 1.f / sm;
      for (int l = 0; l < sz; ++l) ss[t][l] *= inv;
    }
    __syncthreads();
    for (int idx = t; idx < sz * 128; idx += 128) {
      int j = idx >> 7, d = idx & 127;
      float o = 0.f;
      for (int l = 0; l < sz; ++l) o += ss[j][l] * sv[l][d];
      out[(size_t)((estart + j) * 4 + b) * 256 + h * 128 + d] = o;
    }
    __syncthreads();
  }
}

// ---------------------------------------------------------------------------
// Edge cross-attention: mem mask allows exactly keys {0 (z), grp+1}.
// grid = (LE, B), block = 128.
// ---------------------------------------------------------------------------
__global__ void k_edge_cross(const float* __restrict__ qb, const float* __restrict__ kvb,
                             float* __restrict__ out) {
  int e = blockIdx.x, b = blockIdx.y, t = threadIdx.x;
  int grp, nidx; edge_decode(e, grp, nidx);
  size_t row = (size_t)(e * 4 + b);
  size_t m0 = (size_t)b;                 // mem row 0 (z)
  size_t m1 = (size_t)((grp + 1) * 4 + b);
  __shared__ float r0[128], r1[128];
  for (int h = 0; h < 2; ++h) {
    float qv = qb[row * 256 + h * 128 + t];
    r0[t] = qv * kvb[m0 * 512 + h * 128 + t];
    r1[t] = qv * kvb[m1 * 512 + h * 128 + t];
    __syncthreads();
    for (int off = 64; off; off >>= 1) {
      if (t < off) { r0[t] += r0[t + off]; r1[t] += r1[t + off]; }
      __syncthreads();
    }
    float s0 = r0[0] * RSQ_DH, s1 = r1[0] * RSQ_DH;
    float mx = fmaxf(s0, s1);
    float p0 = __expf(s0 - mx), p1 = __expf(s1 - mx);
    float inv = 1.f / (p0 + p1);
    out[row * 256 + h * 128 + t] =
        (p0 * kvb[m0 * 512 + 256 + h * 128 + t] +
         p1 * kvb[m1 * 512 + 256 + h * 128 + t]) * inv;
    __syncthreads();
  }
}

// Edge target gather: tgtE[e,b,:] = node_emb[IDX[e],b,:]
__global__ void k_gather(const float* __restrict__ ne, float* __restrict__ tgtE) {
  int i = blockIdx.x * 256 + threadIdx.x;
  if (i >= LEB * 256) return;
  int d = i & 255, r = i >> 8, e = r >> 2, b = r & 3;
  int grp, nidx; edge_decode(e, grp, nidx);
  tgtE[i] = ne[(size_t)(nidx * 4 + b) * 256 + d];
}

// Edge memory: [z ; node_emb]  -> 1028 rows.
__global__ void k_build_mem(const float* __restrict__ z, const float* __restrict__ ne,
                            float* __restrict__ mem) {
  int i = blockIdx.x * 256 + threadIdx.x;
  if (i >= 1028 * 256) return;
  int d = i & 255, r = i >> 8, m = r >> 2, b = r & 3;
  mem[i] = (m == 0) ? z[b * 256 + d] : ne[(size_t)((m - 1) * 4 + b) * 256 + d];
}

// ---------------------------------------------------------------------------
// Cross-entropy head: one block per row; accumulates nll & valid count.
// ---------------------------------------------------------------------------
__global__ void k_ce(const float* __restrict__ emb, const float* __restrict__ w,
                     const float* __restrict__ bias, const int* __restrict__ y,
                     int nY, int comp, int C, float* __restrict__ acc, int slot) {
  int row = blockIdx.x, t = threadIdx.x;
  __shared__ float sl[64];
  float l = -1e30f;
  if (t < C) {
    const float* wr = w + t * 256;
    const float* er = emb + (size_t)row * 256;
    float a = bias[t];
    for (int d = 0; d < 256; ++d) a += er[d] * wr[d];
    l = a;
  }
  sl[t] = l;
  __syncthreads();
  if (t == 0) {
    float mx = -1e30f;
    for (int c = 0; c < C; ++c) mx = fmaxf(mx, sl[c]);
    float sm = 0.f;
    for (int c = 0; c < C; ++c) sm += __expf(sl[c] - mx);
    float lse = mx + __logf(sm);
    int tg = y[row * nY + comp];
    if (tg >= 0) {
      int tc = tg < C ? tg : C - 1;
      atomicAdd(&acc[2 * slot], lse - sl[tc]);
      atomicAdd(&acc[2 * slot + 1], 1.0f);
    }
  }
}

__global__ void k_zero(float* acc) { if (threadIdx.x < 16) acc[threadIdx.x] = 0.f; }

__global__ void k_final(const float* __restrict__ acc, float* __restrict__ out) {
  if (threadIdx.x == 0) {
    float s = 0.f;
    for (int i = 0; i < 5; ++i) s += acc[2 * i] / fmaxf(acc[2 * i + 1], 1.0f);
    out[0] = s;
  }
}

// ---------------------------------------------------------------------------
// Orchestration
// ---------------------------------------------------------------------------
extern "C" void kernel_launch(void* const* d_in, const int* in_sizes, int n_in,
                              void* d_out, int out_size, void* d_ws, size_t ws_size,
                              hipStream_t stream) {
  (void)in_sizes; (void)n_in; (void)out_size; (void)ws_size;
  const float* z      = (const float*)d_in[0];
  const int*   atom_i = (const int*)d_in[1];
  const float* atom_x = (const float*)d_in[2];
  const int*   atom_y = (const int*)d_in[3];
  const int*   bond_y = (const int*)d_in[4];
  const float* Wp   = (const float*)d_in[5];
  const float* bp   = (const float*)d_in[6];
  const float* bn_g = (const float*)d_in[7];
  const float* bn_b = (const float*)d_in[8];
  // P[0..15]=nd_*, P[16..31]=ed_* in order:
  // qkv_w,qkv_b,so_w,so_b,cqkv_w,cqkv_b,co_w,co_b,ff1_w,ff1_b,ff2_w,ff2_b,ln_g,ln_b,fn_g,fn_b
  const float* P[32];
  for (int i = 0; i < 32; ++i) P[i] = (const float*)d_in[9 + i];
  const float* const* E = P + 16;
  const float* ncw[3] = {(const float*)d_in[41], (const float*)d_in[43], (const float*)d_in[45]};
  const float* ncb[3] = {(const float*)d_in[42], (const float*)d_in[44], (const float*)d_in[46]};
  const float* ecw[2] = {(const float*)d_in[47], (const float*)d_in[49]};
  const float* ecb[2] = {(const float*)d_in[48], (const float*)d_in[50]};
  const int nC[3] = {40, 8, 6};
  const int eC[2] = {5, 4};

  // Workspace layout (floats), ~83 MB total.
  float* ws = (float*)d_ws;
  size_t o = 0;
  float* acc   = ws;       o += 16;
  float* x0    = ws + o;   o += (size_t)1024 * 256;
  float* tgtN  = ws + o;   o += (size_t)1024 * 256;
  float* stats = ws + o;   o += 512;
  float* BZ    = ws + o;   o += 16 * 256;
  float* BZ2   = ws + o;   o += 16 * 256;
  float* tgtE  = ws + o;   o += (size_t)LEB * 256;
  float* memB  = ws + o;   o += (size_t)1028 * 256;
  float* KV    = ws + o;   o += (size_t)1028 * 512;
  float* BQKV  = ws + o;   o += (size_t)LEB * 768;
  float* BA    = ws + o;   o += (size_t)LEB * 256;
  float* BO    = ws + o;   o += (size_t)LEB * 256;
  _Float16* h16 = (_Float16*)(ws + o);   // f16 weight cache (~2M halves)

  k_zero<<<1, 32, 0, stream>>>(acc);

  // ---- One-shot f32->f16 conversion of all GEMM weight matrices ----
  const float* cvs[12] = {P[0], P[2], P[4],  P[6],  P[8],  P[10],
                          E[0], E[2], E[4],  E[6],  E[8],  E[10]};
  const size_t cvn[12] = {2u*768*256, 2u*256*256, 2u*768*256, 2u*256*256,
                          2u*256*256, 2u*256*256,
                          768u*256,   256u*256,   768u*256,   256u*256,
                          256u*256,   256u*256};
  size_t hoff[12]; size_t hco = 0;
  for (int i = 0; i < 12; ++i) {
    hoff[i] = hco;
    k_cvt<<<((int)cvn[i] + 255) / 256, 256, 0, stream>>>(cvs[i], h16 + hco, (int)cvn[i]);
    hco += cvn[i];
  }

  // ---- Input embedding ----
  k_proj<<<1024, 256, 0, stream>>>(atom_x, Wp, bp, x0);
  k_bn_stats<<<256, 256, 0, stream>>>(x0, stats);
  k_bn_selu_pe<<<1024, 256, 0, stream>>>(x0, stats, bn_g, bn_b, atom_i, tgtN);

  // ---- Node decoder: 2 layers ----
  for (int l = 0; l < 2; ++l) {
    const _Float16* WQKV = h16 + hoff[0] + (size_t)l * 768 * 256;
    const _Float16* WSO  = h16 + hoff[1] + (size_t)l * 65536;
    const _Float16* WCV  = h16 + hoff[2] + (size_t)l * 768 * 256 + (size_t)512 * 256;
    const _Float16* WCO  = h16 + hoff[3] + (size_t)l * 65536;
    const _Float16* WF1  = h16 + hoff[4] + (size_t)l * 65536;
    const _Float16* WF2  = h16 + hoff[5] + (size_t)l * 65536;

    gemm(stream, tgtN, WQKV, P[1] + l * 768, BQKV, 1024, 768, 0);
    k_node_self<<<dim3(256, 8), 256, 0, stream>>>(BQKV, BA);
    gemm(stream, BA, WSO, P[3] + l * 256, BO, 1024, 256, 0);
    k_res_ln<<<1024, 256, 0, stream>>>(tgtN, BO, 0, P[12] + (l * 3 + 0) * 256,
                                       P[13] + (l * 3 + 0) * 256, tgtN);
    // cross-attn over single memory token z: softmax == 1 -> out = Wo(V(z))
    gemm(stream, z, WCV, P[5] + l * 768 + 512, BZ, 4, 256, 0);
    gemm(stream, BZ, WCO, P[7] + l * 256, BZ2, 4, 256, 0);
    k_res_ln<<<1024, 256, 0, stream>>>(tgtN, BZ2, 1, P[12] + (l * 3 + 1) * 256,
                                       P[13] + (l * 3 + 1) * 256, tgtN);
    gemm(stream, tgtN, WF1, P[9] + l * 256, BA, 1024, 256, 1);
    gemm(stream, BA, WF2, P[11] + l * 256, BO, 1024, 256, 0);
    k_res_ln<<<1024, 256, 0, stream>>>(tgtN, BO, 0, P[12] + (l * 3 + 2) * 256,
                                       P[13] + (l * 3 + 2) * 256, tgtN);
  }
  k_res_ln<<<1024, 256, 0, stream>>>(tgtN, nullptr, 0, P[14], P[15], tgtN);

  // ---- Node losses ----
  for (int i = 0; i < 3; ++i)
    k_ce<<<1024, 64, 0, stream>>>(tgtN, ncw[i], ncb[i], atom_y, 3, i, nC[i], acc, i);

  // ---- Edge decoder: 1 layer ----
  k_gather<<<(LEB * 256 + 255) / 256, 256, 0, stream>>>(tgtN, tgtE);
  k_build_mem<<<(1028 * 256 + 255) / 256, 256, 0, stream>>>(z, tgtN, memB);
  gemm(stream, tgtE, h16 + hoff[6], E[1], BQKV, LEB, 768, 0);
  k_edge_self<<<dim3(255, 4), 128, 0, stream>>>(BQKV, BA);
  gemm(stream, BA, h16 + hoff[7], E[3], BO, LEB, 256, 0);
  k_res_ln<<<LEB, 256, 0, stream>>>(tgtE, BO, 0, E[12] + 0 * 256, E[13] + 0 * 256, tgtE);
  gemm(stream, tgtE, h16 + hoff[8], E[5], BQKV, LEB, 256, 0);                     // Q
  gemm(stream, memB, h16 + hoff[8] + (size_t)256 * 256, E[5] + 256, KV, 1028, 512, 0); // K|V
  k_edge_cross<<<dim3(LE, 4), 128, 0, stream>>>(BQKV, KV, BA);
  gemm(stream, BA, h16 + hoff[9], E[7], BO, LEB, 256, 0);
  k_res_ln<<<LEB, 256, 0, stream>>>(tgtE, BO, 0, E[12] + 1 * 256, E[13] + 1 * 256, tgtE);
  gemm(stream, tgtE, h16 + hoff[10], E[9], BA, LEB, 256, 1);
  gemm(stream, BA, h16 + hoff[11], E[11], BO, LEB, 256, 0);
  k_res_ln<<<LEB, 256, 0, stream>>>(tgtE, BO, 0, E[12] + 2 * 256, E[13] + 2 * 256, tgtE);
  k_res_ln<<<LEB, 256, 0, stream>>>(tgtE, nullptr, 0, E[14], E[15], tgtE);

  // ---- Edge losses + finalize ----
  for (int i = 0; i < 2; ++i)
    k_ce<<<LEB, 64, 0, stream>>>(tgtE, ecw[i], ecb[i], bond_y, 2, i, eC[i], acc, 3 + i);
  k_final<<<1, 32, 0, stream>>>(acc, (float*)d_out);
}
